// AdaptiveSparseAttention_82600811037214
// MI455X (gfx1250) — matrix operand order repaired and checked by reading, not compiled, not run on hardware
//
#include <hip/hip_runtime.h>
#include <math.h>

typedef __attribute__((ext_vector_type(16))) _Float16 v16h;
typedef __attribute__((ext_vector_type(8)))  float    v8f;

union AFrag { v16h v; uint4 u[2]; _Float16 h[16]; };
union Acc   { v8f  v; float f[8]; };

constexpr int   kB  = 4, kL = 1024, kD = 512, kH = 8, kHd = 64;
constexpr float kScale = 0.125f;    // hd^-0.5
constexpr float kNeg   = -1.0e9f;
constexpr float kNegH  = -60000.0f; // f16-representable "minus infinity" for LDS scores

constexpr int SROWS  = 64;          // query rows per attention block
constexpr int SSTR_H = 1040;        // padded f16 LDS row stride (halves): bank shift 8/row

__device__ __forceinline__ v8f wmma16(v16h a, v16h b, v8f c) {
    return __builtin_amdgcn_wmma_f32_16x16x32_f16(
        /*neg_a=*/false, a, /*neg_b=*/false, b,
        /*c_mod=*/(short)0, c, /*reuse_a=*/false, /*reuse_b=*/false);
}

// A fragment, 16x32 f16 tile, row-major base (16 rows, leading dim lda).
// ISA 7.12.2: lane m=lane&15 holds row m, K = ks..ks+7 and ks+16..ks+23 (ks = 0/8 by lane half).
// Works for both global and LDS (generic) pointers.
__device__ __forceinline__ v16h load_a16(const _Float16* A, int lda, int k0, int lane) {
    int m = lane & 15, ks = (lane >> 4) * 8;
    const _Float16* p = A + (size_t)m * lda + k0 + ks;
    AFrag r;
    r.u[0] = *reinterpret_cast<const uint4*>(p);
    r.u[1] = *reinterpret_cast<const uint4*>(p + 16);
    return r.v;
}

// B fragment from B^T stored row-major (N x K): lane holds column n = lane&15,
// K = ks..ks+7 and ks+16..ks+23 (ks = 0/8 by lane half), matching load_a16.
__device__ __forceinline__ v16h load_bt16(const _Float16* Bt, int ldb, int n0, int k0, int lane) {
    int n = lane & 15, ks = (lane >> 4) * 8;
    const _Float16* p = Bt + (size_t)(n0 + n) * ldb + k0 + ks;
    AFrag r;
    r.u[0] = *reinterpret_cast<const uint4*>(p);
    r.u[1] = *reinterpret_cast<const uint4*>(p + 16);
    return r.v;
}

// ---------------- elementwise f32 -> f16 ----------------
__global__ __launch_bounds__(256) void f32_to_f16(const float* __restrict__ in,
                                                  _Float16* __restrict__ out, int n) {
    int i = blockIdx.x * 256 + threadIdx.x;
    if (i < n) out[i] = (_Float16)in[i];
}

// ---------------- generic WMMA GEMM: C = A(MxK) * Bt(NxK)^T + bias ----------------
// block = 256 threads = 8 waves; wave tile 16x64; block tile 64x128.
__global__ __launch_bounds__(256) void gemm_f16_wmma(const _Float16* __restrict__ A,
                                                     const _Float16* __restrict__ Bt,
                                                     const float* __restrict__ bias,
                                                     float* __restrict__ C,
                                                     int M, int N, int K) {
    int tid = threadIdx.x, lane = tid & 31, wave = tid >> 5;
    int wm = wave & 3, wn = wave >> 2;
    int m0 = blockIdx.x * 64 + wm * 16;
    int n0 = blockIdx.y * 128 + wn * 64;

    Acc acc[4];
    #pragma unroll
    for (int j = 0; j < 4; ++j)
        #pragma unroll
        for (int i = 0; i < 8; ++i) acc[j].f[i] = 0.f;

    const _Float16* Ab = A + (size_t)m0 * K;
    for (int k0 = 0; k0 < K; k0 += 32) {
        // prefetch K tile two ahead (speculative; dropped if OOB) -> global_prefetch_b8
        __builtin_prefetch(Ab + (size_t)(lane & 15) * K + k0 + 64, 0, 0);
        __builtin_prefetch(Bt + (size_t)(n0 + (lane & 15)) * K + k0 + 64, 0, 0);
        v16h a = load_a16(Ab, K, k0, lane);
        #pragma unroll
        for (int j = 0; j < 4; ++j) {
            v16h b = load_bt16(Bt, K, n0 + j * 16, k0, lane);
            acc[j].v = wmma16(a, b, acc[j].v);
        }
    }

    int half = lane >> 4, col0 = lane & 15;
    #pragma unroll
    for (int j = 0; j < 4; ++j) {
        int col = n0 + j * 16 + col0;
        float bv = bias ? bias[col] : 0.f;
        #pragma unroll
        for (int i = 0; i < 8; ++i) {
            int row = m0 + i + half * 8;   // C layout: VGPR i <-> M=i (+8 for hi lane half)
            C[(size_t)row * N + col] = acc[j].f[i] + bv;
        }
    }
}

// ---------------- repack qkv f32 (BL x 1536) -> per-head f16 q,k and v^T ----------------
__global__ __launch_bounds__(256) void repack_qkv(const float* __restrict__ qkv,
                                                  _Float16* __restrict__ qh,
                                                  _Float16* __restrict__ kh,
                                                  _Float16* __restrict__ vTh) {
    int idx = blockIdx.x * 256 + threadIdx.x;      // over 4096*512
    int lg = idx >> 9;                             // global row 0..4095
    int c  = idx & 511;                            // 0..511 = h*64+d
    int b = lg >> 10, l = lg & 1023;
    int h = c >> 6,  d = c & 63;
    const float* base = qkv + (size_t)lg * 1536;
    int bh = b * kH + h;
    qh [((size_t)bh * kL + l) * kHd + d] = (_Float16)base[c];
    kh [((size_t)bh * kL + l) * kHd + d] = (_Float16)base[512 + c];
    vTh[((size_t)bh * kHd + d) * kL + l] = (_Float16)base[1024 + c];
}

// ---------------- fused attention: scores -> softmax -> attn*V ----------------
// grid (L/64, B*H); 256 threads; dynamic LDS: f16 score tile + mask + reduce scratch.
// 64-row blocks halve K/V L2 re-reads vs 32; f16 scores let phase 3 feed WMMA A
// fragments with two ds_load_b128 each (no per-element converts).
__global__ __launch_bounds__(256) void attention_wmma(const _Float16* __restrict__ qh,
                                                      const _Float16* __restrict__ kh,
                                                      const _Float16* __restrict__ vTh,
                                                      const int* __restrict__ attn_mask,
                                                      _Float16* __restrict__ oh) {
    extern __shared__ char smem_raw[];
    _Float16* sc  = (_Float16*)smem_raw;                            // [SROWS][SSTR_H] f16
    int*   smask  = (int*)(smem_raw + (size_t)SROWS * SSTR_H * 2);  // [kL]
    float* sred   = (float*)(smask + kL);                           // [SROWS][4]
    float* srow   = sred + SROWS * 4;                               // [SROWS]

    int tid = threadIdx.x, lane = tid & 31, wave = tid >> 5;
    int rb = blockIdx.x;            // row block (64 query rows)
    int bh = blockIdx.y;            // b*H + h
    int b = bh >> 3, h = bh & 7;
    int half = lane >> 4, col0 = lane & 15;

    for (int c = tid; c < kL; c += 256) smask[c] = attn_mask[b * kL + c];
    __syncthreads();

    // ---- phase 1: scores = mask(clamp(q k^T * scale)) into f16 LDS ----
    {
        int rt = wave & 3;          // row tile within the 64 rows
        int wq = wave >> 2;         // 0..1, each covers 32 column tiles
        const _Float16* qbase = qh + ((size_t)bh * kL + rb * 64 + rt * 16) * kHd;
        const _Float16* kbase = kh + (size_t)bh * kL * kHd;
        v16h a0 = load_a16(qbase, kHd, 0,  lane);
        v16h a1 = load_a16(qbase, kHd, 32, lane);
        for (int i = 0; i < 32; ++i) {
            int ct = wq * 32 + i;
            Acc acc;
            #pragma unroll
            for (int r = 0; r < 8; ++r) acc.f[r] = 0.f;
            v16h b0 = load_bt16(kbase, kHd, ct * 16, 0,  lane);
            v16h b1 = load_bt16(kbase, kHd, ct * 16, 32, lane);
            acc.v = wmma16(a0, b0, acc.v);
            acc.v = wmma16(a1, b1, acc.v);
            int col = ct * 16 + col0;
            bool keep = (smask[col] != 0);
            #pragma unroll
            for (int r = 0; r < 8; ++r) {
                float s = acc.f[r] * kScale;
                // reference clamps at +-1e9; tighten to f16-safe +-60000 (identical
                // softmax result: real |scores| << 60000, and exp beyond underflows)
                s = fminf(fmaxf(s, kNegH), -kNegH);
                if (!keep) s = kNegH;           // padding mask
                sc[(size_t)(rt * 16 + r + half * 8) * SSTR_H + col] = (_Float16)s;
            }
        }
    }
    __syncthreads();

    // ---- phase 2: softmax per row (4 threads per row, 64 rows) ----
    {
        int row = tid >> 2, sub = tid & 3;
        _Float16* prow = sc + (size_t)row * SSTR_H;
        float mx = kNeg;
        for (int c = sub; c < kL; c += 4) mx = fmaxf(mx, (float)prow[c]);
        sred[row * 4 + sub] = mx;
        __syncthreads();
        if (sub == 0) {
            float m = sred[row * 4];
            #pragma unroll
            for (int j = 1; j < 4; ++j) m = fmaxf(m, sred[row * 4 + j]);
            srow[row] = m;
        }
        __syncthreads();
        float rm = srow[row], sum = 0.f;
        for (int c = sub; c < kL; c += 4) {
            float e = __expf((float)prow[c] - rm);
            prow[c] = (_Float16)e;              // exp in [0,1]: safe in f16
            sum += e;
        }
        sred[row * 4 + sub] = sum;
        __syncthreads();
        if (sub == 0) {
            float t = 0.f;
            #pragma unroll
            for (int j = 0; j < 4; ++j) t += sred[row * 4 + j];
            srow[row] = 1.0f / t;               // fold normalization into phase 3 store
        }
        __syncthreads();
    }

    // ---- phase 3: out = (exp_scores @ V) * inv_rowsum -> (B,L,D) f16 ----
    {
        int rt = wave & 3, cv2 = wave >> 2;     // each wave: row tile rt, col tiles cv2*2+{0,1}
        const _Float16* arow = sc + (size_t)(rt * 16) * SSTR_H;   // 16-row A base in LDS
        #pragma unroll
        for (int j = 0; j < 2; ++j) {
            int cv = cv2 * 2 + j;
            const _Float16* vb = vTh + ((size_t)bh * kHd + cv * 16) * kL;
            Acc acc;
            #pragma unroll
            for (int i = 0; i < 8; ++i) acc.f[i] = 0.f;
            for (int kt = 0; kt < kL; kt += 32) {
                v16h a  = load_a16(arow, SSTR_H, kt, lane);   // two ds_load_b128
                v16h bf = load_bt16(vb, kL, 0, kt, lane);
                acc.v = wmma16(a, bf, acc.v);
            }
            int dcol = h * 64 + cv * 16 + col0;
            #pragma unroll
            for (int i = 0; i < 8; ++i) {
                int rloc = rt * 16 + i + half * 8;
                int l = rb * 64 + rloc;
                oh[((size_t)b * kL + l) * kD + dcol] = (_Float16)(acc.f[i] * srow[rloc]);
            }
        }
    }
}

extern "C" void kernel_launch(void* const* d_in, const int* in_sizes, int n_in,
                              void* d_out, int out_size, void* d_ws, size_t ws_size,
                              hipStream_t stream) {
    const float* x         = (const float*)d_in[0];
    const int*   attn_mask = (const int*)  d_in[1];
    const float* qkv_w     = (const float*)d_in[2];
    const float* proj_w    = (const float*)d_in[3];
    const float* proj_b    = (const float*)d_in[4];
    // d_in[5..10] (selector MLP, sparse_w/b) are mathematically dead:
    // pw = softmax(.) > 0 componentwise, so combined = pw0*local + pw1 + pw2*sparse > 0
    // everywhere -> the soft-OR mask and the top-k sparse mask never mask anything.

    char* ws = (char*)d_ws;
    size_t off = 0;
    auto alloc = [&](size_t bytes) {
        char* p = ws + off;
        off = (off + bytes + 255) & ~(size_t)255;
        return p;
    };
    const int BL = kB * kL;                                           // 4096
    _Float16* x_h     = (_Float16*)alloc((size_t)BL * kD * 2);        // 4 MB
    _Float16* qkvw_h  = (_Float16*)alloc((size_t)3 * kD * kD * 2);    // 1.5 MB
    _Float16* projw_h = (_Float16*)alloc((size_t)kD * kD * 2);        // 0.5 MB
    float*    qkv_f   = (float*)   alloc((size_t)BL * 3 * kD * 4);    // 25 MB
    _Float16* qh      = (_Float16*)alloc((size_t)BL * kD * 2);
    _Float16* kh      = (_Float16*)alloc((size_t)BL * kD * 2);
    _Float16* vTh     = (_Float16*)alloc((size_t)BL * kD * 2);
    _Float16* oh      = (_Float16*)alloc((size_t)BL * kD * 2);

    int nx = BL * kD, nqw = 3 * kD * kD, npw = kD * kD;
    f32_to_f16<<<(nx  + 255) / 256, 256, 0, stream>>>(x,      x_h,     nx);
    f32_to_f16<<<(nqw + 255) / 256, 256, 0, stream>>>(qkv_w,  qkvw_h,  nqw);
    f32_to_f16<<<(npw + 255) / 256, 256, 0, stream>>>(proj_w, projw_h, npw);

    // QKV projection: (4096x512) * (1536x512)^T
    gemm_f16_wmma<<<dim3(BL / 64, (3 * kD) / 128), 256, 0, stream>>>(
        x_h, qkvw_h, nullptr, qkv_f, BL, 3 * kD, kD);

    repack_qkv<<<(BL * kD) / 256, 256, 0, stream>>>(qkv_f, qh, kh, vTh);

    size_t shbytes = (size_t)SROWS * SSTR_H * 2   // f16 scores
                   + (size_t)kL * 4               // mask
                   + (size_t)SROWS * 4 * 4        // partial reductions
                   + (size_t)SROWS * 4;           // row max / inv-sum
    attention_wmma<<<dim3(kL / SROWS, kB * kH), 256, shbytes, stream>>>(
        qh, kh, vTh, attn_mask, oh);

    // output projection: (4096x512) * (512x512)^T + bias -> f32 d_out
    gemm_f16_wmma<<<dim3(BL / 64, kD / 128), 256, 0, stream>>>(
        oh, projw_h, proj_b, (float*)d_out, BL, kD, kD);

    (void)in_sizes; (void)n_in; (void)out_size; (void)ws_size;
}
